// GNNLowerBound_47132971106610
// MI455X (gfx1250) — matrix operand order repaired and checked by reading, not compiled, not run on hardware
//
#include <hip/hip_runtime.h>
#include <math.h>

// ============================================================================
// MI455X (gfx1250) fused GNN forward.
//  - wave32, 16-row tiles per wave, v_wmma_f32_16x16x32_f16 everywhere.
//  - per-edge matmuls hoisted to per-node; edge pipeline fully fused
//    (gather + LN + WMMA + f32 atomic scatter-add segment_sum).
//  - weights pre-packed into WMMA B-fragment-major f16 layout: one 32B vector
//    load per fragment instead of 16 strided u16 loads + packing movs.
//  - node tables (12.8MB) live in the 192MB L2 -> gathers are L2 hits.
// ============================================================================

typedef __attribute__((ext_vector_type(16))) _Float16 v16h;
typedef __attribute__((ext_vector_type(8)))  float    v8f;

#define WPB 8            // waves per block
#define TB  (WPB * 32)   // threads per block

// ---------------------------------------------------------------------------
// WMMA helpers (CDNA5 16x16x32 f16 -> f32)
// ---------------------------------------------------------------------------
__device__ __forceinline__ v8f wmma(v16h a, v16h b, v8f c) {
  return __builtin_amdgcn_wmma_f32_16x16x32_f16(false, a, false, b, (short)0, c,
                                                false, false);
}

// Pre-packed B fragment: frag id f = k2*4 + nt; lane's 16 halves contiguous.
__device__ __forceinline__ v16h bfragP(const _Float16* __restrict__ Wp, int k2,
                                       int nt, int lane) {
  const v16h* p = (const v16h*)Wp;
  return p[(k2 * 4 + nt) * 32 + lane];
}

// A-fragment column owned by this lane: index i in [0,32) over both k-steps.
// col = (i/16)*32 + ((i/8)&1)*16 + hi*8 + (i&7)
__device__ __forceinline__ int lane_col(int i, int hi) {
  return ((i >> 4) << 5) + (((i >> 3) & 1) << 4) + (hi << 3) + (i & 7);
}

// Load the 32 f32 values of one 64-float row that this lane owns (A layout).
__device__ __forceinline__ void load_row_cols(const float* __restrict__ row,
                                              int hi, float v[32]) {
#pragma unroll
  for (int s = 0; s < 2; ++s) {
#pragma unroll
    for (int p = 0; p < 2; ++p) {
      const float4* src = (const float4*)(row + s * 32 + p * 16 + hi * 8);
      float4 x0 = src[0], x1 = src[1];
      float* d = v + s * 16 + p * 8;
      d[0] = x0.x; d[1] = x0.y; d[2] = x0.z; d[3] = x0.w;
      d[4] = x1.x; d[5] = x1.y; d[6] = x1.z; d[7] = x1.w;
    }
  }
}

__device__ __forceinline__ void frags_from_vals(const float v[32], v16h& a0,
                                                v16h& a1) {
#pragma unroll
  for (int j = 0; j < 16; ++j) {
    a0[j] = (_Float16)v[j];
    a1[j] = (_Float16)v[16 + j];
  }
}

// ---------------------------------------------------------------------------
// Weight pre-pack: f32 W[K x 64] -> f16 fragment-major:
//   dst[((k2*4+nt)*32 + lane)*16 + j] = W[(k2*32 + (j>>3)*16 + (lane>>4)*8 + (j&7))*64
//                                         + nt*16 + (lane&15)]
// ---------------------------------------------------------------------------
__global__ void k_pack(const float* __restrict__ s, _Float16* __restrict__ d,
                       int K) {
  int tid = blockIdx.x * blockDim.x + threadIdx.x;
  if (tid >= K * 64) return;
  const int j    = tid & 15;
  const int lane = (tid >> 4) & 31;
  const int f    = tid >> 9;
  const int k2 = f >> 2, nt = f & 3;
  const int k = k2 * 32 + ((j >> 3) << 4) + ((lane >> 4) << 3) + (j & 7);
  const int n = nt * 16 + (lane & 15);
  d[tid] = (_Float16)s[k * 64 + n];
}

// ---------------------------------------------------------------------------
// Fused embedding: LN(din) -> Linear(din,64)+ReLU -> WMMA(64,64)+ReLU
// ---------------------------------------------------------------------------
__global__ void k_emb(const float* __restrict__ x, int din,
                      const float* __restrict__ lng, const float* __restrict__ lnb,
                      const float* __restrict__ W1, const float* __restrict__ b1,
                      const _Float16* __restrict__ W2, const float* __restrict__ b2,
                      int N, float* __restrict__ outF, _Float16* __restrict__ outH) {
  const int lane = threadIdx.x & 31, hi = lane >> 4, r = lane & 15;
  const int tile = blockIdx.x * WPB + (threadIdx.x >> 5);
  const int base = tile * 16;
  if (base >= N) return;
  const int row = base + r;

  // tiny-dim LayerNorm, computed redundantly by both lanes of the row pair
  float xin[3]; float mean = 0.f;
#pragma unroll
  for (int d = 0; d < 3; ++d) {
    float t = (d < din) ? x[(size_t)row * din + d] : 0.f;
    xin[d] = t; mean += t;
  }
  mean /= (float)din;
  float var = 0.f;
#pragma unroll
  for (int d = 0; d < 3; ++d)
    if (d < din) { float c = xin[d] - mean; var += c * c; }
  var /= (float)din;
  const float inv = rsqrtf(var + 1e-5f);
  float xl[3];
#pragma unroll
  for (int d = 0; d < 3; ++d)
    xl[d] = (d < din) ? ((xin[d] - mean) * inv * lng[d] + lnb[d]) : 0.f;

  // hidden1 at this lane's A-fragment columns (vectorized param loads)
  float vb1[32];
  load_row_cols(b1, hi, vb1);
  float vW1[3][32];
#pragma unroll
  for (int d = 0; d < 3; ++d) {
    if (d < din) load_row_cols(W1 + d * 64, hi, vW1[d]);
    else {
#pragma unroll
      for (int i = 0; i < 32; ++i) vW1[d][i] = 0.f;
    }
  }
  float v[32];
#pragma unroll
  for (int i = 0; i < 32; ++i) {
    float h = vb1[i];
#pragma unroll
    for (int d = 0; d < 3; ++d) h += xl[d] * vW1[d][i];
    v[i] = fmaxf(h, 0.f);
  }
  v16h a0, a1; frags_from_vals(v, a0, a1);

#pragma unroll
  for (int nt = 0; nt < 4; ++nt) {
    v8f acc = {};
    acc = wmma(a0, bfragP(W2, 0, nt, lane), acc);
    acc = wmma(a1, bfragP(W2, 1, nt, lane), acc);
    const int n = nt * 16 + r;
    const float bn = b2[n];
    float y[8];
#pragma unroll
    for (int vv = 0; vv < 8; ++vv) y[vv] = fmaxf(acc[vv] + bn, 0.f);
    if (outF) {
#pragma unroll
      for (int vv = 0; vv < 8; ++vv)
        outF[(size_t)(base + hi * 8 + vv) * 64 + n] = y[vv];
    }
    if (outH) {
#pragma unroll
      for (int vv = 0; vv < 8; ++vv)
        outH[(size_t)(base + hi * 8 + vv) * 64 + n] = (_Float16)y[vv];
    }
  }
}

// ---------------------------------------------------------------------------
// Generic Y = act(X[N,64] @ W[64,64] + b)
// ---------------------------------------------------------------------------
__global__ void k_gemm64(const float* __restrict__ X, const _Float16* __restrict__ W,
                         const float* __restrict__ bias, int relu, int N,
                         float* __restrict__ outF) {
  const int lane = threadIdx.x & 31, hi = lane >> 4, r = lane & 15;
  const int tile = blockIdx.x * WPB + (threadIdx.x >> 5);
  const int base = tile * 16;
  if (base >= N) return;

  float v[32];
  load_row_cols(X + (size_t)(base + r) * 64, hi, v);
  v16h a0, a1; frags_from_vals(v, a0, a1);

#pragma unroll
  for (int nt = 0; nt < 4; ++nt) {
    v8f acc = {};
    acc = wmma(a0, bfragP(W, 0, nt, lane), acc);
    acc = wmma(a1, bfragP(W, 1, nt, lane), acc);
    const int n = nt * 16 + r;
    const float bn = bias ? bias[n] : 0.f;
#pragma unroll
    for (int vv = 0; vv < 8; ++vv) {
      float y = acc[vv] + bn;
      if (relu) y = fmaxf(y, 0.f);
      outF[(size_t)(base + hi * 8 + vv) * 64 + n] = y;
    }
  }
}

// ---------------------------------------------------------------------------
// Fused edge pipeline:
//   m = PR[dst] + PL[src] + (e16 @ We); y = relu(LN(m))@Wf + bf;
//   agg[dst] += y  (global f32 atomics == segment_sum)
// Wave-private LDS tile; same-wave DS ordering (no block barriers).
// ---------------------------------------------------------------------------
__global__ void k_edge(const _Float16* __restrict__ e16,
                       const int* __restrict__ srcI, const int* __restrict__ dstI,
                       const float* __restrict__ PR, const float* __restrict__ PL,
                       const _Float16* __restrict__ We,
                       const float* __restrict__ fg, const float* __restrict__ fb,
                       const _Float16* __restrict__ Wf, const float* __restrict__ bf,
                       float* __restrict__ agg, int E) {
  __shared__ float tile_s[WPB][16][68];
  __shared__ int   dsts[WPB][16];
  const int wid = threadIdx.x >> 5;
  const int lane = threadIdx.x & 31, hi = lane >> 4, r = lane & 15;
  const int t = blockIdx.x * WPB + wid;
  const int base = t * 16;
  if (base >= E) return;
  float (&T)[16][68] = tile_s[wid];

  const int eid = base + r;
  const int di = dstI[eid];
  const int si = srcI[eid];
  if (hi == 0) dsts[wid][r] = di;

  // A-fragments of f16 edge-embedding row (contiguous 16B chunks)
  const _Float16* er = e16 + (size_t)eid * 64;
  v16h e0, e1;
#pragma unroll
  for (int i = 0; i < 32; ++i) {
    _Float16 h = er[lane_col(i, hi)];
    if (i < 16) e0[i] = h; else e1[i - 16] = h;
  }
  // PE = e @ We -> LDS (D layout -> row-major tile)
#pragma unroll
  for (int nt = 0; nt < 4; ++nt) {
    v8f acc = {};
    acc = wmma(e0, bfragP(We, 0, nt, lane), acc);
    acc = wmma(e1, bfragP(We, 1, nt, lane), acc);
    const int n = nt * 16 + r;
#pragma unroll
    for (int vv = 0; vv < 8; ++vv) T[hi * 8 + vv][n] = acc[vv];
  }
  asm volatile("s_wait_dscnt 0x0" ::: "memory");  // wave-local LDS ordering

  // gathered node projections + PE, then LayerNorm stats over the row pair
  float vg[32];
  load_row_cols(PR + (size_t)di * 64, hi, vg);
  {
    float vl[32];
    load_row_cols(PL + (size_t)si * 64, hi, vl);
#pragma unroll
    for (int i = 0; i < 32; ++i) vg[i] += vl[i];
  }
  float s = 0.f, ss = 0.f;
#pragma unroll
  for (int i = 0; i < 32; ++i) {
    vg[i] += T[r][lane_col(i, hi)];
    s += vg[i]; ss += vg[i] * vg[i];
  }
  s  += __shfl_xor(s, 16, 32);
  ss += __shfl_xor(ss, 16, 32);
  const float mean = s * (1.f / 64.f);
  const float var  = ss * (1.f / 64.f) - mean * mean;
  const float inv  = rsqrtf(var + 1e-5f);

  float vfg[32], vfb[32];
  load_row_cols(fg, hi, vfg);
  load_row_cols(fb, hi, vfb);
  v16h m0, m1;
#pragma unroll
  for (int i = 0; i < 32; ++i) {
    float y = fmaxf((vg[i] - mean) * inv * vfg[i] + vfb[i], 0.f);
    if (i < 16) m0[i] = (_Float16)y; else m1[i - 16] = (_Float16)y;
  }
  // y = m @ Wf + bf -> atomic scatter-add into agg[dst]
#pragma unroll
  for (int nt = 0; nt < 4; ++nt) {
    v8f acc = {};
    acc = wmma(m0, bfragP(Wf, 0, nt, lane), acc);
    acc = wmma(m1, bfragP(Wf, 1, nt, lane), acc);
    const int n = nt * 16 + r;
    const float bn = bf[n];
#pragma unroll
    for (int vv = 0; vv < 8; ++vv) {
      const int dd = dsts[wid][hi * 8 + vv];
      __hip_atomic_fetch_add(&agg[(size_t)dd * 64 + n], acc[vv] + bn,
                             __ATOMIC_RELAXED, __HIP_MEMORY_SCOPE_AGENT);
    }
  }
}

// ---------------------------------------------------------------------------
// Fused post-conv: out = relu([LN(agg), right] @ Wo1 + bo1) @ Wo2 + bo2
// ---------------------------------------------------------------------------
__global__ void k_node(const float* __restrict__ agg, const float* __restrict__ right,
                       const float* __restrict__ pg, const float* __restrict__ pb,
                       const _Float16* __restrict__ Wo1, const float* __restrict__ bo1,
                       const _Float16* __restrict__ Wo2, const float* __restrict__ bo2,
                       int N, float* __restrict__ outF) {
  __shared__ float tile_s[WPB][16][68];
  const int wid = threadIdx.x >> 5;
  const int lane = threadIdx.x & 31, hi = lane >> 4, r = lane & 15;
  const int tile = blockIdx.x * WPB + wid;
  const int base = tile * 16;
  if (base >= N) return;
  float (&T)[16][68] = tile_s[wid];

  float va[32];
  load_row_cols(agg + (size_t)(base + r) * 64, hi, va);
  float s = 0.f, ss = 0.f;
#pragma unroll
  for (int i = 0; i < 32; ++i) { s += va[i]; ss += va[i] * va[i]; }
  s  += __shfl_xor(s, 16, 32);
  ss += __shfl_xor(ss, 16, 32);
  const float mean = s * (1.f / 64.f);
  const float var  = ss * (1.f / 64.f) - mean * mean;
  const float inv  = rsqrtf(var + 1e-5f);

  float vpg[32], vpb[32];
  load_row_cols(pg, hi, vpg);
  load_row_cols(pb, hi, vpb);
  v16h a0, a1;
#pragma unroll
  for (int i = 0; i < 32; ++i) {
    float t = (va[i] - mean) * inv * vpg[i] + vpb[i];
    if (i < 16) a0[i] = (_Float16)t; else a1[i - 16] = (_Float16)t;
  }
  float vr[32];
  load_row_cols(right + (size_t)(base + r) * 64, hi, vr);
  v16h a2, a3; frags_from_vals(vr, a2, a3);

#pragma unroll
  for (int nt = 0; nt < 4; ++nt) {
    v8f acc = {};
    acc = wmma(a0, bfragP(Wo1, 0, nt, lane), acc);
    acc = wmma(a1, bfragP(Wo1, 1, nt, lane), acc);
    acc = wmma(a2, bfragP(Wo1, 2, nt, lane), acc);
    acc = wmma(a3, bfragP(Wo1, 3, nt, lane), acc);
    const int n = nt * 16 + r;
    const float bn = bo1[n];
#pragma unroll
    for (int vv = 0; vv < 8; ++vv)
      T[hi * 8 + vv][n] = fmaxf(acc[vv] + bn, 0.f);
  }
  asm volatile("s_wait_dscnt 0x0" ::: "memory");

  v16h h0, h1;
#pragma unroll
  for (int i = 0; i < 32; ++i) {
    const float t = T[r][lane_col(i, hi)];
    if (i < 16) h0[i] = (_Float16)t; else h1[i - 16] = (_Float16)t;
  }
#pragma unroll
  for (int nt = 0; nt < 4; ++nt) {
    v8f acc = {};
    acc = wmma(h0, bfragP(Wo2, 0, nt, lane), acc);
    acc = wmma(h1, bfragP(Wo2, 1, nt, lane), acc);
    const int n = nt * 16 + r;
    const float bn = bo2[n];
#pragma unroll
    for (int vv = 0; vv < 8; ++vv)
      outF[(size_t)(base + hi * 8 + vv) * 64 + n] = acc[vv] + bn;
  }
}

// ---------------------------------------------------------------------------
// Small utility kernels
// ---------------------------------------------------------------------------
__global__ void k_zero(float* __restrict__ p, int n) {
  int i = blockIdx.x * blockDim.x + threadIdx.x;
  if (i < n) p[i] = 0.f;
}
__global__ void k_rowdot_sig(const float* __restrict__ T, const float* __restrict__ w,
                             float* __restrict__ out, int N) {
  int i = blockIdx.x * blockDim.x + threadIdx.x;
  if (i >= N) return;
  const float* row = T + (size_t)i * 64;
  float acc = 0.f;
#pragma unroll
  for (int c = 0; c < 64; ++c) acc += row[c] * w[c];
  out[i] = 1.f / (1.f + expf(-acc));
}
__global__ void k_rowdot_sum(const float* __restrict__ T, const float* __restrict__ w,
                             float* __restrict__ out, int N) {
  __shared__ float red[256];
  int i = blockIdx.x * 256 + threadIdx.x;
  float acc = 0.f;
  if (i < N) {
    const float* row = T + (size_t)i * 64;
#pragma unroll
    for (int c = 0; c < 64; ++c) acc += row[c] * w[c];
  }
  red[threadIdx.x] = acc;
  __syncthreads();
  for (int s2 = 128; s2 > 0; s2 >>= 1) {
    if (threadIdx.x < s2) red[threadIdx.x] += red[threadIdx.x + s2];
    __syncthreads();
  }
  if (threadIdx.x == 0)
    __hip_atomic_fetch_add(out, red[0], __ATOMIC_RELAXED, __HIP_MEMORY_SCOPE_AGENT);
}

// ---------------------------------------------------------------------------
// Host orchestration
// ---------------------------------------------------------------------------
extern "C" void kernel_launch(void* const* d_in, const int* in_sizes, int n_in,
                              void* d_out, int out_size, void* d_ws, size_t ws_size,
                              hipStream_t stream) {
  (void)n_in; (void)out_size; (void)ws_size;
  const float* antF = (const float*)d_in[0];
  const float* edgF = (const float*)d_in[1];
  const float* usrF = (const float*)d_in[2];
  const int*   eix  = (const int*)d_in[3];
  const int NA = in_sizes[0] / 2;
  const int E  = in_sizes[1] / 3;
  const int NU = in_sizes[2];
  const int* srcA = eix;       // edge_index[0] : antenna ids
  const int* dstU = eix + E;   // edge_index[1] : user ids

  // param leaves in setup_inputs() insertion order
#define IN(i) ((const float*)d_in[(i)])
  // emb_a:4..9  emb_e:10..15  emb_u:16..21
  // a2u: 22 Wl,23 bl,24 We,25 Wr,26 flng,27 flnb,28 Wf,29 bf,30 plng,31 plnb,
  //      32 Wo1,33 bo1,34 Wo2,35 bo2
  // u2a: 36..49 same layout; 50 Wi1,51 bi1,52 Wi2,53 Wp1,54 bp1,55 Wp2

  // ---- workspace bump allocator -------------------------------------------
  char* ws = (char*)d_ws;
  size_t off = 0;
  auto a16 = [&](size_t n) -> _Float16* {
    off = (off + 255) & ~(size_t)255; _Float16* r = (_Float16*)(ws + off);
    off += n * sizeof(_Float16); return r;
  };
  auto a32 = [&](size_t n) -> float* {
    off = (off + 255) & ~(size_t)255; float* r = (float*)(ws + off);
    off += n * sizeof(float); return r;
  };

  _Float16 *W2a = a16(4096), *W2e = a16(4096), *W2u = a16(4096);
  _Float16 *Wl0 = a16(4096), *We0 = a16(4096), *Wr0 = a16(4096), *Wf0 = a16(4096);
  _Float16 *Wo1_0 = a16(8192), *Wo2_0 = a16(4096);
  _Float16 *Wl1 = a16(4096), *We1 = a16(4096), *Wr1 = a16(4096), *Wf1 = a16(4096);
  _Float16 *Wo1_1 = a16(8192), *Wo2_1 = a16(4096);
  _Float16 *Wi1h = a16(4096), *Wp1h = a16(4096);

  const int NM = NA > NU ? NA : NU;
  float    *a0  = a32((size_t)NA * 64), *u0 = a32((size_t)NU * 64);
  _Float16 *e16 = a16((size_t)E * 64);
  float    *PRb = a32((size_t)NM * 64), *PLb = a32((size_t)NM * 64);
  float    *agg = a32((size_t)NM * 64);
  float    *u1  = a32((size_t)NU * 64), *a1 = a32((size_t)NA * 64);
  float    *fub = a32((size_t)NU * 64), *tmp = a32((size_t)NM * 64);

  auto grid16 = [](int n) { int t = (n + 15) / 16; return (t + WPB - 1) / WPB; };
  auto pack = [&](const float* s, _Float16* d, int K) {
    k_pack<<<(K * 64 + 255) / 256, 256, 0, stream>>>(s, d, K);
  };
  auto zero = [&](float* p, int n) {
    k_zero<<<(n + 255) / 256, 256, 0, stream>>>(p, n);
  };

  // ---- f16 fragment-packed weight conversion ------------------------------
  pack(IN(8),  W2a, 64); pack(IN(14), W2e, 64); pack(IN(20), W2u, 64);
  pack(IN(22), Wl0, 64); pack(IN(24), We0, 64); pack(IN(25), Wr0, 64);
  pack(IN(28), Wf0, 64); pack(IN(32), Wo1_0, 128); pack(IN(34), Wo2_0, 64);
  pack(IN(36), Wl1, 64); pack(IN(38), We1, 64); pack(IN(39), Wr1, 64);
  pack(IN(42), Wf1, 64); pack(IN(46), Wo1_1, 128); pack(IN(48), Wo2_1, 64);
  pack(IN(50), Wi1h, 64); pack(IN(53), Wp1h, 64);

  // ---- embeddings ---------------------------------------------------------
  k_emb<<<grid16(NA), TB, 0, stream>>>(antF, 2, IN(4), IN(5), IN(6), IN(7),
                                       W2a, IN(9), NA, a0, nullptr);
  k_emb<<<grid16(E),  TB, 0, stream>>>(edgF, 3, IN(10), IN(11), IN(12), IN(13),
                                       W2e, IN(15), E, nullptr, e16);
  k_emb<<<grid16(NU), TB, 0, stream>>>(usrF, 1, IN(16), IN(17), IN(18), IN(19),
                                       W2u, IN(21), NU, u0, nullptr);

  // ---- conv1: a2u (left=a0, right=u0, agg over users) ---------------------
  k_gemm64<<<grid16(NU), TB, 0, stream>>>(u0, Wl0, IN(23), 0, NU, PRb);
  k_gemm64<<<grid16(NA), TB, 0, stream>>>(a0, Wr0, nullptr, 0, NA, PLb);
  zero(agg, NU * 64);
  k_edge<<<grid16(E), TB, 0, stream>>>(e16, srcA, dstU, PRb, PLb, We0,
                                       IN(26), IN(27), Wf0, IN(29), agg, E);
  k_node<<<grid16(NU), TB, 0, stream>>>(agg, u0, IN(30), IN(31), Wo1_0, IN(33),
                                        Wo2_0, IN(35), NU, u1);

  // ---- conv2: u2a (left=u1, right=a0, reversed edges, agg over antennas) --
  k_gemm64<<<grid16(NA), TB, 0, stream>>>(a0, Wl1, IN(37), 0, NA, PRb);
  k_gemm64<<<grid16(NU), TB, 0, stream>>>(u1, Wr1, nullptr, 0, NU, PLb);
  zero(agg, NA * 64);
  k_edge<<<grid16(E), TB, 0, stream>>>(e16, dstU, srcA, PRb, PLb, We1,
                                       IN(40), IN(41), Wf1, IN(43), agg, E);
  k_node<<<grid16(NA), TB, 0, stream>>>(agg, a0, IN(44), IN(45), Wo1_1, IN(47),
                                        Wo2_1, IN(49), NA, a1);

  // ---- conv3: a2u again, shared weights (left=a1, right=u1) ---------------
  k_gemm64<<<grid16(NU), TB, 0, stream>>>(u1, Wl0, IN(23), 0, NU, PRb);
  k_gemm64<<<grid16(NA), TB, 0, stream>>>(a1, Wr0, nullptr, 0, NA, PLb);
  zero(agg, NU * 64);
  k_edge<<<grid16(E), TB, 0, stream>>>(e16, srcA, dstU, PRb, PLb, We0,
                                       IN(26), IN(27), Wf0, IN(29), agg, E);
  k_node<<<grid16(NU), TB, 0, stream>>>(agg, u1, IN(30), IN(31), Wo1_0, IN(33),
                                        Wo2_0, IN(35), NU, fub);

  // ---- heads --------------------------------------------------------------
  float* outI = (float*)d_out;
  k_gemm64<<<grid16(NA), TB, 0, stream>>>(a1, Wi1h, IN(51), 1, NA, tmp);
  k_rowdot_sig<<<(NA + 255) / 256, 256, 0, stream>>>(tmp, IN(52), outI, NA);

  zero(outI + NA, 1);
  k_gemm64<<<grid16(NU), TB, 0, stream>>>(fub, Wp1h, IN(54), 1, NU, tmp);
  k_rowdot_sum<<<(NU + 255) / 256, 256, 0, stream>>>(tmp, IN(55), outI + NA, NU);
#undef IN
}
